// SpeakingEncoder_56504589746623
// MI455X (gfx1250) — matrix operand order repaired and checked by reading, not compiled, not run on hardware
//
#include <hip/hip_runtime.h>
#include <hip/hip_bf16.h>

// ---------------------------------------------------------------------------
// VQ encoder for MI455X (gfx1250, wave32, WMMA).
//  x:        [8, 2048, 1024] f32   (d_in[0])
//  codebook: [4096, 1024]    f32   (d_in[1])
//  out:      [8, 2048, 1024] f32 = codebook[argmin_n dist(x, c_n)] + PE(s, d)
// argmin_n (||x||^2 - 2 x.c + ||c||^2) == argmin_n (||c||^2 - 2 x.c)
// ---------------------------------------------------------------------------

typedef __attribute__((ext_vector_type(16))) __bf16 v16bf;
typedef __attribute__((ext_vector_type(8)))  __bf16 v8bf;
typedef __attribute__((ext_vector_type(4)))  __bf16 v4bf;
typedef __attribute__((ext_vector_type(8)))  float  v8f;

#define D_MODEL       1024
#define N_CODES       4096
#define N_TOK         16384      // 8 * 2048
#define SEQ_LEN       2048
#define M_TILE        64         // tokens per workgroup (4 row-groups x 16)
#define N_SPLIT       4          // codebook split across blockIdx.y
#define N_PER_SPLIT   (N_CODES / N_SPLIT)   // 1024
#define LDS_ROW_PITCH 1032       // 1024 + 8 bf16 pad -> 516-dword row stride (bank-conflict free)

// --------------------------- kernel 1: codebook prep -----------------------
// fp32 codebook -> bf16 copy (for WMMA B operand) + squared row norms.
__global__ void vq_prep_codebook(const float* __restrict__ cb,
                                 __bf16* __restrict__ cbB,
                                 float* __restrict__ cnorm) {
  __shared__ float red[8];
  const int row = blockIdx.x;
  const int tid = threadIdx.x;
  const int j   = tid * 4;                       // 256 thr * 4 = 1024 elems
  float4 f = *(const float4*)(cb + (size_t)row * D_MODEL + j);
  v4bf h;
  h[0] = (__bf16)f.x; h[1] = (__bf16)f.y; h[2] = (__bf16)f.z; h[3] = (__bf16)f.w;
  *(v4bf*)(cbB + (size_t)row * D_MODEL + j) = h;
  float s = f.x * f.x + f.y * f.y + f.z * f.z + f.w * f.w;
#pragma unroll
  for (int m = 16; m >= 1; m >>= 1) s += __shfl_xor(s, m, 32);
  if ((tid & 31) == 0) red[tid >> 5] = s;
  __syncthreads();
  if (tid == 0) {
    float t = 0.0f;
#pragma unroll
    for (int w = 0; w < 8; ++w) t += red[w];
    cnorm[row] = t;
  }
}

// --------------- kernel 2: fused bf16 WMMA GEMM + running argmin -----------
// grid = (N_TOK / M_TILE, N_SPLIT), block = 256 (8 wave32).
// Waves 0-3 and 4-7 own the same 4x16-row slabs of the x tile but disjoint
// halves of the block's N range; per 64-wide N panel each wave accumulates a
// 16x64 score panel (4 x v_wmma_f32_16x16x32_bf16 per K-step, software-
// pipelined with register-double-buffered A/B fragments) over K=1024, then
// folds (||c||^2 - 2 x.c) into per-lane running minima.
__global__ __launch_bounds__(256, 2)
void vq_gemm_argmin(const float* __restrict__ x,
                    const __bf16* __restrict__ cbB,
                    const float* __restrict__ cnorm,
                    unsigned long long* __restrict__ keys) {
  extern __shared__ __bf16 ldsx[];               // M_TILE * LDS_ROW_PITCH bf16
  const int tid  = threadIdx.x;
  const int lane = tid & 31;
  const int wave = tid >> 5;                     // 0..7
  const int wrow = wave & 3;                     // row-group within tile
  const int wseg = wave >> 2;                    // N-segment within split
  const int half = lane >> 4;                    // 0/1
  const int l15  = lane & 15;

  // ---- stage the 64x1024 x tile into LDS as bf16 (coalesced float4) ----
  const float* xblk = x + (size_t)blockIdx.x * M_TILE * D_MODEL;
  for (int i = tid * 4; i < M_TILE * D_MODEL; i += 256 * 4) {
    float4 f = *(const float4*)(xblk + i);
    const int row = i >> 10;
    const int k   = i & (D_MODEL - 1);
    v4bf h;
    h[0] = (__bf16)f.x; h[1] = (__bf16)f.y; h[2] = (__bf16)f.z; h[3] = (__bf16)f.w;
    *(v4bf*)(&ldsx[row * LDS_ROW_PITCH + k]) = h;
  }
  __syncthreads();

  // A-fragment lane mapping (16-bit A 16x32): lane l15 holds row M=l15;
  // lanes 0-15 carry K {0..15}, lanes 16-31 carry K {16..31}.
  const __bf16* aBase = &ldsx[(wrow * 16 + l15) * LDS_ROW_PITCH];

  float runval[8];
  int   runidx[8];
#pragma unroll
  for (int r = 0; r < 8; ++r) { runval[r] = 3.4e38f; runidx[r] = 0; }

  const int nsplit0 = blockIdx.y * N_PER_SPLIT;
  for (int n0 = nsplit0 + wseg * 64; n0 < nsplit0 + N_PER_SPLIT; n0 += 128) {
    v8f acc[4] = {};
    // B-fragment (16-bit B 32x16): lane holds column N=l15, contiguous K run;
    // lanes 0-15 -> K[k0..k0+15], lanes 16-31 -> K[k0+16..k0+31].
    // One base pointer; the per-tile offsets t*16*D_MODEL (32 KB) fold into
    // the load instruction's signed-24-bit immediate -> no extra VGPRs.
    const __bf16* bb = cbB + (size_t)(n0 + l15) * D_MODEL + (half << 4);

    // -------- software pipeline: prologue loads for k0 = 0 --------
    v8bf plo = *(const v8bf*)(aBase + half * 16);
    v8bf phi = *(const v8bf*)(aBase + 8 + half * 16);
    v16bf aCur = __builtin_shufflevector(plo, phi,
                     0, 1, 2, 3, 4, 5, 6, 7, 8, 9, 10, 11, 12, 13, 14, 15);
    v16bf bCur[4];
#pragma unroll
    for (int t = 0; t < 4; ++t)
      bCur[t] = *(const v16bf*)(bb + t * (16 * D_MODEL));

#pragma unroll 2
    for (int k0 = 0; k0 < D_MODEL - 32; k0 += 32) {
      const int kn = k0 + 32;
      // issue next-step fragment loads before this step's WMMAs
      v8bf nlo = *(const v8bf*)(aBase + kn + half * 16);
      v8bf nhi = *(const v8bf*)(aBase + kn + 8 + half * 16);
      v16bf aNxt = __builtin_shufflevector(nlo, nhi,
                       0, 1, 2, 3, 4, 5, 6, 7, 8, 9, 10, 11, 12, 13, 14, 15);
      v16bf bNxt[4];
#pragma unroll
      for (int t = 0; t < 4; ++t)
        bNxt[t] = *(const v16bf*)(bb + kn + t * (16 * D_MODEL));
      __builtin_prefetch((const void*)(bb + k0 + 128), 0, 3);
      __builtin_prefetch((const void*)(bb + k0 + 128 + 2 * (16 * D_MODEL)), 0, 3);
#pragma unroll
      for (int t = 0; t < 4; ++t) {
        acc[t] = __builtin_amdgcn_wmma_f32_16x16x32_bf16(
            /*neg_a=*/false, aCur, /*neg_b=*/false, bCur[t],
            /*c_mod=*/(short)0, acc[t], /*reuse_a=*/false, /*reuse_b=*/false);
      }
      aCur = aNxt;
#pragma unroll
      for (int t = 0; t < 4; ++t) bCur[t] = bNxt[t];
    }
    // -------- epilogue: last K step --------
#pragma unroll
    for (int t = 0; t < 4; ++t) {
      acc[t] = __builtin_amdgcn_wmma_f32_16x16x32_bf16(
          false, aCur, false, bCur[t], (short)0, acc[t], false, false);
    }

    // D layout (f32 16x16): VGPR r, lanes 0-15 -> (M=r, N=l15),
    // lanes 16-31 -> (M=8+r, N=l15). Fold into running argmin.
#pragma unroll
    for (int t = 0; t < 4; ++t) {
      const int ncol = n0 + t * 16 + l15;
      const float cn = cnorm[ncol];
#pragma unroll
      for (int r = 0; r < 8; ++r) {
        const float score = cn - 2.0f * acc[t][r];
        if (score < runval[r]) { runval[r] = score; runidx[r] = ncol; }
      }
    }
  }

  // Reduce the 16 column-slots (lexicographic: value, then smaller index ==
  // first occurrence, matching jnp.argmin), then merge across N partitions
  // with a packed-key u64 atomic min (order-preserving float bits | index).
  const size_t tokbase = (size_t)blockIdx.x * M_TILE + wrow * 16 + half * 8;
#pragma unroll
  for (int r = 0; r < 8; ++r) {
    float v = runval[r];
    int   i = runidx[r];
#pragma unroll
    for (int m = 1; m <= 8; m <<= 1) {
      const float ov = __shfl_xor(v, m, 32);
      const int   oi = __shfl_xor(i, m, 32);
      if (ov < v || (ov == v && oi < i)) { v = ov; i = oi; }
    }
    if (l15 == 0) {
      unsigned ub = __float_as_uint(v);
      ub = (ub & 0x80000000u) ? ~ub : (ub | 0x80000000u);
      const unsigned long long key =
          ((unsigned long long)ub << 32) | (unsigned)i;
      atomicMin(&keys[tokbase + r], key);
    }
  }
}

// ----------------- kernel 3: gather codebook row + positional enc ----------
__global__ void vq_gather_pe(const float* __restrict__ cb,
                             const unsigned long long* __restrict__ keys,
                             float* __restrict__ out) {
  const int token = blockIdx.x;
  const int s     = token & (SEQ_LEN - 1);
  const unsigned idx = (unsigned)(keys[token] & 0xFFFFFFFFull);
  const int d0 = threadIdx.x * 4;                // even; pairs (d0,d0+1),(d0+2,d0+3)
  float4 c = *(const float4*)(cb + (size_t)idx * D_MODEL + d0);
  const float LOG_OVER_D = 9.210340371976184f / 1024.0f;   // ln(10000)/D
  const float div0 = __expf(-(float)d0 * LOG_OVER_D);
  const float div1 = __expf(-(float)(d0 + 2) * LOG_OVER_D);
  const float a0 = (float)s * div0;
  const float a1 = (float)s * div1;
  float4 o;
  o.x = c.x + sinf(a0);
  o.y = c.y + cosf(a0);
  o.z = c.z + sinf(a1);
  o.w = c.w + cosf(a1);
  *(float4*)(out + (size_t)token * D_MODEL + d0) = o;
}

// ---------------------------------------------------------------------------
extern "C" void kernel_launch(void* const* d_in, const int* in_sizes, int n_in,
                              void* d_out, int out_size, void* d_ws, size_t ws_size,
                              hipStream_t stream) {
  const float* x  = (const float*)d_in[0];     // [8,2048,1024]
  const float* cb = (const float*)d_in[1];     // [4096,1024]
  char* ws = (char*)d_ws;

  const size_t cbB_bytes   = (size_t)N_CODES * D_MODEL * sizeof(__bf16); // 8 MB
  const size_t cnorm_bytes = (size_t)N_CODES * sizeof(float);            // 16 KB
  __bf16* cbB  = (__bf16*)ws;
  float*  cnorm = (float*)(ws + cbB_bytes);
  unsigned long long* keys = (unsigned long long*)(ws + cbB_bytes + cnorm_bytes);

  // keys = +inf score / index 0xFFFFFFFF sentinel
  hipMemsetAsync(keys, 0xFF, (size_t)N_TOK * sizeof(unsigned long long), stream);

  vq_prep_codebook<<<N_CODES, 256, 0, stream>>>(cb, cbB, cnorm);

  const int lds_bytes = M_TILE * LDS_ROW_PITCH * (int)sizeof(__bf16);   // 129 KB -> 2 WGs/WGP
  (void)hipFuncSetAttribute(reinterpret_cast<const void*>(vq_gemm_argmin),
                            hipFuncAttributeMaxDynamicSharedMemorySize, lds_bytes);
  dim3 grid2(N_TOK / M_TILE, N_SPLIT);
  vq_gemm_argmin<<<grid2, 256, lds_bytes, stream>>>(x, cbB, cnorm, keys);

  vq_gather_pe<<<N_TOK, 256, 0, stream>>>(cb, keys, (float*)d_out);
}
